// CGWeight_4870492914238
// MI455X (gfx1250) — matrix operand order repaired and checked by reading, not compiled, not run on hardware
//
#include <hip/hip_runtime.h>

typedef __attribute__((ext_vector_type(16))) _Float16 v16h;
typedef __attribute__((ext_vector_type(8)))  float    v8f;
typedef __attribute__((ext_vector_type(2)))  _Float16 half2v;

// ---------------------------------------------------------------------------
// Setup: compute real SO(3) CG coefficients (e3nn convention) on device in
// f64, fold in weight[c], emit f16 B-matrix already swizzled into the WMMA
// 16-bit B-fragment layout:  Bfrag[chunk c][lane][vgpr v] dword, halves hh:
//   K = 32*c + (lane>=16 ? 8 : 0) + (v>=4 ? 16 : 0) + 2*(v&3) + hh,  n = lane & 15
//   B[K][n] = (n<5 && combo valid) ? weight[c]*GC[i=K>>4][j=K&15][n] : 0
// ---------------------------------------------------------------------------

struct cxd { double re, im; };
__device__ inline cxd cmul(cxd a, cxd b) { return {a.re*b.re - a.im*b.im, a.re*b.im + a.im*b.re}; }

__device__ inline double factd(int n) {
    const double F[13] = {1.0, 1.0, 2.0, 6.0, 24.0, 120.0, 720.0, 5040.0,
                          40320.0, 362880.0, 3628800.0, 39916800.0, 479001600.0};
    return F[n];
}

__device__ double su2_coeff(int j1, int m1, int j2, int m2, int j3, int m3) {
    if (m3 != m1 + m2) return 0.0;
    int vmin = -j1 + j2 + m3; if (-j1 + m1 > vmin) vmin = -j1 + m1; if (vmin < 0) vmin = 0;
    int vmax = j2 + j3 + m1; if (j3 - j1 + j2 < vmax) vmax = j3 - j1 + j2; if (j3 + m3 < vmax) vmax = j3 + m3;
    double C = sqrt((2.0*j3 + 1.0) *
                    factd(j3+j1-j2) * factd(j3-j1+j2) * factd(j1+j2-j3) *
                    factd(j3+m3) * factd(j3-m3) /
                    (factd(j1+j2+j3+1) * factd(j1-m1) * factd(j1+m1) *
                     factd(j2-m2) * factd(j2+m2)));
    double S = 0.0;
    for (int v = vmin; v <= vmax; ++v) {
        double term = factd(j2+j3+m1-v) * factd(j1-m1+v) /
                      (factd(v) * factd(j3-j1+j2-v) * factd(j3+m3-v) * factd(v+j1-j2-m3));
        S += ((v + j2 + m2) & 1) ? -term : term;
    }
    return C * S;
}

// Q (real->complex SH basis) entry, row = l+m (0..2l), col (0..2l), times (-i)^l
__device__ cxd q_entry(int l, int row, int col) {
    const double is2 = 0.70710678118654752440;
    int m = row - l;
    cxd v = {0.0, 0.0};
    if (m < 0) {
        if (col == l - m)      v = {is2, 0.0};    // col == l + |m|
        else if (col == l + m) v = {0.0, -is2};   // col == l - |m|
    } else if (m == 0) {
        if (col == l) v = {1.0, 0.0};
    } else {
        double s = (m & 1) ? -1.0 : 1.0;
        if (col == l + m)      v = {s * is2, 0.0};
        else if (col == l - m) v = {0.0, s * is2};
    }
    switch (l & 3) {                 // multiply by (-i)^l
        case 0: break;
        case 1: v = { v.im, -v.re}; break;
        case 2: v = {-v.re, -v.im}; break;
        case 3: v = {-v.im,  v.re}; break;
    }
    return v;
}

// Real part of einsum('ij,kl,mn,ikn->jlm', Q1, Q2, conj(Q3^T), C)[p,q,r];
// Q columns have <=2 nonzero rows, so <=8 terms.
__device__ double so3_elem(int l1, int l2, int l3, int p, int q, int r) {
    int iset[2] = {p, 2*l1 - p};
    int kset[2] = {q, 2*l2 - q};
    int nset[2] = {r, 2*l3 - r};
    int ni = (iset[1] == iset[0]) ? 1 : 2;
    int nk = (kset[1] == kset[0]) ? 1 : 2;
    int nn = (nset[1] == nset[0]) ? 1 : 2;
    double inv_norm = 1.0 / sqrt(2.0*l3 + 1.0);
    double acc = 0.0;
    for (int a = 0; a < ni; ++a) {
        int i = iset[a];
        cxd q1 = q_entry(l1, i, p);
        if (q1.re == 0.0 && q1.im == 0.0) continue;
        for (int b = 0; b < nk; ++b) {
            int k = kset[b];
            cxd q2 = q_entry(l2, k, q);
            if (q2.re == 0.0 && q2.im == 0.0) continue;
            cxd q12 = cmul(q1, q2);
            for (int cc = 0; cc < nn; ++cc) {
                int n = nset[cc];
                int m1 = i - l1, m2 = k - l2, m3 = n - l3;
                if (m3 != m1 + m2) continue;
                cxd q3 = q_entry(l3, n, r);
                cxd q3c = {q3.re, -q3.im};
                double cval = su2_coeff(l1, m1, l2, m2, l3, m3) * inv_norm;
                cxd t = cmul(q12, q3c);
                acc += t.re * cval;
            }
        }
    }
    return acc;
}

__device__ inline void blockmap(int g, int& l, int& loc) {
    if (g == 0)      { l = 0; loc = 0; }
    else if (g < 4)  { l = 1; loc = g - 1; }
    else if (g < 9)  { l = 2; loc = g - 4; }
    else             { l = 3; loc = g - 9; }
}

__global__ void setup_bfrag(const float* __restrict__ weight, _Float16* __restrict__ Bws) {
    // combo index table in (ai outer, hi inner) enumeration order; -1 = invalid
    const int tbl[16] = {-1, -1, 0, -1,
                         -1,  1, 2,  3,
                          4,  5, 6,  7,
                         -1,  8, 9, 10};
    int d = blockIdx.x * blockDim.x + threadIdx.x;   // 0..2047 dwords
    if (d >= 2048) return;
    int c    = d >> 8;
    int lane = (d >> 3) & 31;
    int v    = d & 7;
    for (int hh = 0; hh < 2; ++hh) {
        int K = 32*c + ((lane & 16) ? 8 : 0) + ((v & 4) ? 16 : 0) + 2*(v & 3) + hh;
        int n = lane & 15;
        int i = K >> 4, j = K & 15;
        float val = 0.0f;
        if (n < 5) {
            int la, ia, lh, jh;
            blockmap(i, la, ia);
            blockmap(j, lh, jh);
            int cidx = tbl[la * 4 + lh];
            if (cidx >= 0) {
                double gc = so3_elem(la, lh, 2, ia, jh, n);
                val = (float)((double)weight[cidx] * gc);
            }
        }
        Bws[d * 2 + hh] = (_Float16)val;
    }
}

// ---------------------------------------------------------------------------
// Main kernel: one 16-edge tile per wave iteration.
//   P(16 x 256) @ B(256 x 16, cols 5..15 zero) via 8x v_wmma_f32_16x16x32_f16.
// Lane pair (q, q+16) owns edge base+q; lane<16 builds j in [0,8) products,
// lane>=16 builds j in [8,16) products per the 16-bit A-fragment layout.
// ---------------------------------------------------------------------------
__global__ __launch_bounds__(256) void cg_wmma_kernel(
    const float* __restrict__ a0, const float* __restrict__ a1,
    const float* __restrict__ a2, const float* __restrict__ a3,
    const float* __restrict__ h0, const float* __restrict__ h1,
    const float* __restrict__ h2p, const float* __restrict__ h3,
    const _Float16* __restrict__ Bws, float* __restrict__ out, int N)
{
    const int lane = threadIdx.x & 31;
    const int hiHalf = (lane & 16) ? 1 : 0;
    const int q = lane & 15;
    const int jb = hiHalf ? 8 : 0;

    const int wavesPerBlock = blockDim.x >> 5;
    const long wave   = (long)blockIdx.x * wavesPerBlock + (threadIdx.x >> 5);
    const long nWaves = (long)gridDim.x * wavesPerBlock;

    // Load constant B fragments once per wave (L2-resident, 256B/lane).
    const v16h* Bf = (const v16h*)Bws;
    v16h bfrag[8];
#pragma unroll
    for (int c = 0; c < 8; ++c) bfrag[c] = Bf[c * 32 + lane];

    const long numTiles = ((long)N + 15) >> 4;

    for (long tile = wave; tile < numTiles; tile += nWaves) {
        const long base = tile << 4;
        long e = base + q;
        long ec = (e < (long)N) ? e : (long)N - 1;   // clamp: keep EXEC full for WMMA

        float af[16], hf[16];
        af[0] = a0[ec];
#pragma unroll
        for (int t = 0; t < 3; ++t) af[1 + t] = a1[ec * 3 + t];
#pragma unroll
        for (int t = 0; t < 5; ++t) af[4 + t] = a2[ec * 5 + t];
#pragma unroll
        for (int t = 0; t < 7; ++t) af[9 + t] = a3[ec * 7 + t];
        hf[0] = h0[ec];
#pragma unroll
        for (int t = 0; t < 3; ++t) hf[1 + t] = h1[ec * 3 + t];
#pragma unroll
        for (int t = 0; t < 5; ++t) hf[4 + t] = h2p[ec * 5 + t];
#pragma unroll
        for (int t = 0; t < 7; ++t) hf[9 + t] = h3[ec * 7 + t];

        _Float16 av[16];
#pragma unroll
        for (int t = 0; t < 16; ++t) av[t] = (_Float16)af[t];
        half2v hj[4];
#pragma unroll
        for (int t = 0; t < 4; ++t) {
            hj[t][0] = (_Float16)hf[jb + 2 * t];
            hj[t][1] = (_Float16)hf[jb + 2 * t + 1];
        }

        v8f acc = {0.f, 0.f, 0.f, 0.f, 0.f, 0.f, 0.f, 0.f};
#pragma unroll
        for (int c = 0; c < 8; ++c) {
            v16h A;
#pragma unroll
            for (int v = 0; v < 8; ++v) {
                const int ii = 2 * c + (v >> 2);
                half2v aa; aa[0] = av[ii]; aa[1] = av[ii];
                half2v pr = aa * hj[v & 3];           // v_pk_mul_f16
                A[2 * v]     = pr[0];
                A[2 * v + 1] = pr[1];
            }
            acc = __builtin_amdgcn_wmma_f32_16x16x32_f16(
                false, A, false, bfrag[c], (short)0, acc, false, false);
        }

        // D layout: VGPR r -> row M=r (lanes 0-15) / M=8+r (lanes 16-31); N = lane&15
        if (q < 5) {
#pragma unroll
            for (int rr = 0; rr < 8; ++rr) {
                long em = base + (hiHalf ? 8 + rr : rr);
                if (em < (long)N) out[em * 5 + q] = acc[rr];
            }
        }
    }
}

extern "C" void kernel_launch(void* const* d_in, const int* in_sizes, int n_in,
                              void* d_out, int out_size, void* d_ws, size_t ws_size,
                              hipStream_t stream) {
    const float* a0 = (const float*)d_in[0];
    const float* a1 = (const float*)d_in[1];
    const float* a2 = (const float*)d_in[2];
    const float* a3 = (const float*)d_in[3];
    const float* h0 = (const float*)d_in[4];
    const float* h1 = (const float*)d_in[5];
    const float* h2 = (const float*)d_in[6];
    const float* h3 = (const float*)d_in[7];
    const float* w  = (const float*)d_in[8];
    float* out = (float*)d_out;
    int N = in_sizes[0];

    _Float16* Bws = (_Float16*)d_ws;   // 8 KB fragment table

    setup_bfrag<<<8, 256, 0, stream>>>(w, Bws);

    int blocks = 1024;                 // 8192 waves, ~15 tiles each at N=2M
    cg_wmma_kernel<<<blocks, 256, 0, stream>>>(a0, a1, a2, a3, h0, h1, h2, h3,
                                               Bws, out, N);
}
